// LayerGCN_61967788146761
// MI455X (gfx1250) — matrix operand order repaired and checked by reading, not compiled, not run on hardware
//
#include <hip/hip_runtime.h>
#include <hip/hip_bf16.h>

typedef __attribute__((ext_vector_type(16))) _Float16 v16h;
typedef __attribute__((ext_vector_type(8)))  _Float16 v8h;
typedef __attribute__((ext_vector_type(8)))  float    v8f;
typedef __attribute__((ext_vector_type(4)))  float    v4f;
typedef __attribute__((ext_vector_type(4)))  unsigned int v4u;

#define THRESH   0.1f
#define BN_EPS   1e-5f
#define N_NODES  128
#define C_IN     64
#define C_HID    256
#define C_OUT    128
#define BLOCK    256

// ws layout (halves/floats):
//  w1sw : 16384 halves (32KB)   swizzled f16 W1
//  w2sw : 32768 halves (64KB)   swizzled f16 W2
//  coef : 768 floats  (3KB)     [al1(256) | be1(256) | al2(128) | be2(128)]
#define WS_W1_HALVES  (C_IN * C_HID)
#define WS_W2_HALVES  (C_HID * C_OUT)
#define WS_COEF_FLOATS (2 * C_HID + 2 * C_OUT)

// ---- A fragment (16x32 f16, MxK), row-major LDS, contiguous k = 16*(lane/16) + s, as in B ------------
__device__ __forceinline__ v16h frag_a(const _Float16* lds, int mBase, int kBase, int ldk) {
    const int lane = threadIdx.x & 31;
    const int m  = mBase + (lane & 15);
    const int kb = kBase + ((lane >> 4) << 4);
    const _Float16* p = lds + m * ldk + kb;
    const v8h lo = *(const v8h*)(p);
    const v8h hi = *(const v8h*)(p + 8);
    return __builtin_shufflevector(lo, hi, 0, 1, 2, 3, 4, 5, 6, 7,
                                           8, 9, 10, 11, 12, 13, 14, 15);
}

// ---- Swizzled B storage: 32x16 (KxN) tiles of 512 halves --------------------
// (k,n) -> tile*512 + lane*16 + (k&15), lane = ((k>>4)&1)*16 + (n&15); fragment
// read is one contiguous 32B per-lane load (2x ds_load_b128, conflict-free).
__device__ __forceinline__ int bsw_idx(int k, int n, int ktiles) {
    const int tile = (n >> 4) * ktiles + (k >> 5);
    const int lane = (((k >> 4) & 1) << 4) + (n & 15);
    return (tile << 9) + (lane << 4) + (k & 15);
}

__device__ __forceinline__ v16h frag_b_sw(const _Float16* lds, int ktile, int ntile, int ktiles) {
    const int lane = threadIdx.x & 31;
    const _Float16* p = lds + (((ntile * ktiles + ktile) << 9) + (lane << 4));
    return *(const v16h*)p;   // 32B contiguous, 32B aligned
}

__device__ __forceinline__ v8f wmma16(v16h a, v16h b, v8f c) {
    return __builtin_amdgcn_wmma_f32_16x16x32_f16(false, a, false, b, (short)0, c, false, false);
}

// ============================================================================
// Prep kernel (runs once per launch): swizzle W1/W2 to f16 and fold BN into
// per-column (alpha, beta).  64 blocks x 256 threads, 1 W1 + 2 W2 elems each.
// ============================================================================
__global__ __launch_bounds__(BLOCK)
void layergcn_prep_kernel(const float* __restrict__ W1, const float* __restrict__ W2,
                          const float* __restrict__ b1, const float* __restrict__ g1,
                          const float* __restrict__ be1, const float* __restrict__ rm1,
                          const float* __restrict__ rv1,
                          const float* __restrict__ b2, const float* __restrict__ g2,
                          const float* __restrict__ be2, const float* __restrict__ rm2,
                          const float* __restrict__ rv2,
                          _Float16* __restrict__ w1sw, _Float16* __restrict__ w2sw,
                          float* __restrict__ coef)
{
    const int i = blockIdx.x * BLOCK + threadIdx.x;           // 0..16383
    { const int k = i >> 8, n = i & 255; w1sw[bsw_idx(k, n, 2)] = (_Float16)W1[i]; }
    { const int k = i >> 7, n = i & 127; w2sw[bsw_idx(k, n, 8)] = (_Float16)W2[i]; }
    { const int j = i + 16384;
      const int k = j >> 7, n = j & 127; w2sw[bsw_idx(k, n, 8)] = (_Float16)W2[j]; }
    if (blockIdx.x == 0) {
        const int t = threadIdx.x;
        const float s1 = rsqrtf(rv1[t] + BN_EPS);
        const float a1 = g1[t] * s1;
        coef[t]       = a1;                                   // alpha1
        coef[256 + t] = (b1[t] - rm1[t]) * a1 + be1[t];       // beta1
        if (t < C_OUT) {
            const float s2 = rsqrtf(rv2[t] + BN_EPS);
            const float a2 = g2[t] * s2;
            coef[512 + t] = a2;                               // alpha2
            coef[640 + t] = (b2[t] - rm2[t]) * a2 + be2[t];   // beta2
        }
    }
}

// ============================================================================
// Main fused kernel: one workgroup per batch.
// ============================================================================
extern __shared__ __align__(64) char smem_raw[];

__global__ __launch_bounds__(BLOCK)
void layergcn_fused_kernel(const float* __restrict__ A,        // [B,128,128]
                           const float* __restrict__ V,        // [B,128,64]
                           const _Float16* __restrict__ w1sw,  // swizzled f16
                           const _Float16* __restrict__ w2sw,  // swizzled f16
                           const float* __restrict__ coef,     // folded BN
                           float* __restrict__ out)            // [B*128,128]
{
    const int tid  = threadIdx.x;
    const int wave = tid >> 5;
    const int lane = tid & 31;
    const size_t b = blockIdx.x;

    // ---- LDS carve-up (single allocation so regions alias contiguously) ----
    _Float16* sA  = (_Float16*)smem_raw;           // 128x128 = 32KB (later W1-sw; W2-sw spans sA..sH)
    _Float16* sV  = sA + N_NODES * N_NODES;        // 128x64  = 16KB (swizzled B)
    _Float16* sH  = sV + N_NODES * C_IN;           // 128x64  = 16KB (row-major A operand)
    _Float16* sX1 = sH + N_NODES * C_IN;           // 128x256 = 64KB (row-major A operand)
    float* sDinv = (float*)(sX1 + N_NODES * C_HID);
    float* sCoef = sDinv + N_NODES;                // al1|be1|al2|be2 = 768 floats
    float* sAl1  = sCoef;
    float* sBe1  = sAl1 + C_HID;
    float* sAl2  = sBe1 + C_HID;
    float* sBe2  = sAl2 + C_OUT;

    // ---- Stage 0: bulk-copy folded BN coeffs (768 floats = 192 x 16B) -------
    if (tid < WS_COEF_FLOATS / 4)
        ((v4u*)sCoef)[tid] = ((const v4u*)coef)[tid];

    // ---- Stage 1: stream A, threshold to binary f16 (row-major, A operand) --
    {
        const v4f* Ab = (const v4f*)(A + b * (size_t)(N_NODES * N_NODES));
#pragma unroll
        for (int i = tid; i < N_NODES * N_NODES / 4; i += BLOCK) {
            const v4f q = __builtin_nontemporal_load(Ab + i);
            sA[4 * i + 0] = (_Float16)(q.x >= THRESH ? 1.0f : 0.0f);
            sA[4 * i + 1] = (_Float16)(q.y >= THRESH ? 1.0f : 0.0f);
            sA[4 * i + 2] = (_Float16)(q.z >= THRESH ? 1.0f : 0.0f);
            sA[4 * i + 3] = (_Float16)(q.w >= THRESH ? 1.0f : 0.0f);
        }
    }
    __syncthreads();

    // ---- Stage 2: degrees -> d^{-1/2} (exact packed-f16 accumulation) -------
    if (tid < N_NODES) {
        const v8h* row = (const v8h*)(sA + tid * N_NODES);
        v8h s8 = row[0];
#pragma unroll
        for (int j = 1; j < N_NODES / 8; ++j) s8 += row[j];   // counts <= 128: exact in f16
        float s = 0.0f;
#pragma unroll
        for (int h = 0; h < 8; ++h) s += (float)s8[h];
        sDinv[tid] = rsqrtf(s);   // matches sum**-0.5 (inf on empty row)
    }
    __syncthreads();

    // ---- Stage 3: V' = d^{-1/2}(row) * V, f16, swizzled B layout (K=128,N=64)
    {
        const v4f* Vb = (const v4f*)(V + b * (size_t)(N_NODES * C_IN));
#pragma unroll
        for (int i = tid; i < N_NODES * C_IN / 4; i += BLOCK) {
            const v4f q = __builtin_nontemporal_load(Vb + i);
            const int k = (4 * i) >> 6;        // row of V = K index, const in vec4
            const int n0 = (4 * i) & 63;       // col of V = N index
            const float dm = sDinv[k];
            sV[bsw_idx(k, n0 + 0, 4)] = (_Float16)(q.x * dm);
            sV[bsw_idx(k, n0 + 1, 4)] = (_Float16)(q.y * dm);
            sV[bsw_idx(k, n0 + 2, 4)] = (_Float16)(q.z * dm);
            sV[bsw_idx(k, n0 + 3, 4)] = (_Float16)(q.w * dm);
        }
    }
    __syncthreads();

    // ---- GEMM1: H = Ath @ V'  (128x64, K=128), row-scale by d^{-1/2} --------
    {
        v16h afr[4];
#pragma unroll
        for (int kk = 0; kk < 4; ++kk) afr[kk] = frag_a(sA, wave * 16, kk * 32, N_NODES);
#pragma unroll
        for (int nT = 0; nT < 4; ++nT) {
            v8f acc = {};
#pragma unroll
            for (int kk = 0; kk < 4; ++kk) acc = wmma16(afr[kk], frag_b_sw(sV, kk, nT, 4), acc);
            const int n  = nT * 16 + (lane & 15);
            const int mb = wave * 16 + ((lane >> 4) << 3);
#pragma unroll
            for (int r = 0; r < 8; ++r) {
                const int m = mb + r;
                sH[m * C_IN + n] = (_Float16)(acc[r] * sDinv[m]);   // row-major (A operand)
            }
        }
    }
    __syncthreads();

    // ---- Bulk-copy pre-swizzled W1 into (dead) sA region (2048 x 16B) -------
    {
        const v4u* src = (const v4u*)w1sw;
        v4u* dst = (v4u*)sA;
#pragma unroll
        for (int i = tid; i < WS_W1_HALVES / 8; i += BLOCK) dst[i] = src[i];
    }
    __syncthreads();

    // ---- GEMM2: X1 = relu(alpha1 * (H @ W1) + beta1)  (128x256, K=64) -------
    {
        v16h afr[2];
#pragma unroll
        for (int kk = 0; kk < 2; ++kk) afr[kk] = frag_a(sH, wave * 16, kk * 32, C_IN);
#pragma unroll
        for (int nT = 0; nT < 16; ++nT) {
            v8f acc = {};
#pragma unroll
            for (int kk = 0; kk < 2; ++kk) acc = wmma16(afr[kk], frag_b_sw(sA, kk, nT, 2), acc);
            const int n  = nT * 16 + (lane & 15);
            const int mb = wave * 16 + ((lane >> 4) << 3);
            const float a1 = sAl1[n], c1 = sBe1[n];
#pragma unroll
            for (int r = 0; r < 8; ++r) {
                const float y = fmaxf(acc[r] * a1 + c1, 0.0f);
                sX1[(mb + r) * C_HID + n] = (_Float16)y;        // row-major (A operand)
            }
        }
    }
    __syncthreads();

    // ---- Bulk-copy pre-swizzled W2 into dead sA+sV+sH span (4096 x 16B) -----
    {
        const v4u* src = (const v4u*)w2sw;
        v4u* dst = (v4u*)sA;
#pragma unroll
        for (int i = tid; i < WS_W2_HALVES / 8; i += BLOCK) dst[i] = src[i];
    }
    __syncthreads();

    // ---- GEMM3: Out = relu(alpha2 * (X1 @ W2) + beta2)  (128x128, K=256) ----
    {
        v16h afr[8];
#pragma unroll
        for (int kk = 0; kk < 8; ++kk) afr[kk] = frag_a(sX1, wave * 16, kk * 32, C_HID);
        float* outb = out + b * (size_t)(N_NODES * C_OUT);
#pragma unroll
        for (int nT = 0; nT < 8; ++nT) {
            v8f acc = {};
#pragma unroll
            for (int kk = 0; kk < 8; ++kk) acc = wmma16(afr[kk], frag_b_sw(sA, kk, nT, 8), acc);
            const int n  = nT * 16 + (lane & 15);
            const int mb = wave * 16 + ((lane >> 4) << 3);
            const float a2 = sAl2[n], c2 = sBe2[n];
#pragma unroll
            for (int r = 0; r < 8; ++r) {
                const float y = fmaxf(acc[r] * a2 + c2, 0.0f);
                __builtin_nontemporal_store(y, outb + (size_t)(mb + r) * C_OUT + n);
            }
        }
    }
}

extern "C" void kernel_launch(void* const* d_in, const int* in_sizes, int n_in,
                              void* d_out, int out_size, void* d_ws, size_t ws_size,
                              hipStream_t stream) {
    const float* A   = (const float*)d_in[0];
    const float* V   = (const float*)d_in[1];
    const float* W1  = (const float*)d_in[2];
    const float* b1  = (const float*)d_in[3];
    const float* g1  = (const float*)d_in[4];
    const float* be1 = (const float*)d_in[5];
    const float* rm1 = (const float*)d_in[6];
    const float* rv1 = (const float*)d_in[7];
    const float* W2  = (const float*)d_in[8];
    const float* b2  = (const float*)d_in[9];
    const float* g2  = (const float*)d_in[10];
    const float* be2 = (const float*)d_in[11];
    const float* rm2 = (const float*)d_in[12];
    const float* rv2 = (const float*)d_in[13];
    float* out = (float*)d_out;

    // workspace carve-up
    _Float16* w1sw = (_Float16*)d_ws;                         // 32KB
    _Float16* w2sw = w1sw + WS_W1_HALVES;                     // 64KB
    float*    coef = (float*)(w2sw + WS_W2_HALVES);           // 3KB

    const int nBatch = in_sizes[0] / (N_NODES * N_NODES);     // 4096

    // Prep: swizzle weights to f16 + fold BN (tiny, L2-resident output)
    layergcn_prep_kernel<<<64, BLOCK, 0, stream>>>(
        W1, W2, b1, g1, be1, rm1, rv1, b2, g2, be2, rm2, rv2, w1sw, w2sw, coef);

    // LDS: A/W (32K) + V (16K) + H (16K) + X1 (64K) + dinv/BN coeffs (~3.5K)
    const size_t shmem =
        (size_t)(N_NODES * N_NODES + N_NODES * C_IN + N_NODES * C_IN + N_NODES * C_HID) * sizeof(_Float16)
        + (N_NODES + WS_COEF_FLOATS) * sizeof(float);

    layergcn_fused_kernel<<<nBatch, BLOCK, shmem, stream>>>(A, V, w1sw, w2sw, coef, out);
}
